// Decoder_37280316129813
// MI455X (gfx1250) — matrix operand order repaired and checked by reading, not compiled, not run on hardware
//
#include <hip/hip_runtime.h>
#include <math.h>

// ---------------- problem constants ----------------
#define BB   128
#define SS   256
#define TT   64
#define HH   512
#define EE   256
#define H2   1024          // 2H
#define KIH  1280          // E + 2H
#define G3   1536          // 3H
#define PP   1792          // H + 2H + E

typedef __attribute__((ext_vector_type(16))) __bf16 v16bf;
typedef __attribute__((ext_vector_type(8)))  __bf16 v8bf;
typedef __attribute__((ext_vector_type(8)))  float  v8f;

// ---------------- fragment load helpers ----------------
// Concatenate two contiguous 8 x bf16 chunks (two b128 loads) into a 16-elem frag.
__device__ __forceinline__ v16bf load16(const __bf16* p0, const __bf16* p1) {
    v8bf a = *(const v8bf*)p0;
    v8bf b = *(const v8bf*)p1;
    v16bf r;
#pragma unroll
    for (int i = 0; i < 8; ++i) { r[i] = a[i]; r[8 + i] = b[i]; }
    return r;
}

__device__ __forceinline__ float sigmoidf_(float x) { return 1.0f / (1.0f + __expf(-x)); }

// ---------------- WMMA GEMM:  C(M,N) = act( A(M,K) @ W(N,K)^T + bias ) ----------------
// A, W bf16 row-major.  One wave computes a 32x64 tile:
//   2 A fragments x 4 B fragments -> 8 wmma accumulators; B frags reused across both
//   M-subtiles (1.5 b128 loads per WMMA).  K-loop is unrolled by 2 with explicit
//   ping-pong fragment buffers: each half loads directly into the register set the
//   other half consumes (no loop-carried copies, no v_mov rotation), so the
//   s_wait_loadcnt before each WMMA block covers loads issued ~8 WMMAs earlier.
// Requires K % 64 == 0 (true for all call sites: 512/1024/1280/1792).
// CDNA5 bf16 operand layout (ISA 7.12.2/7.12.4):
//   A frag, lane half h, row m=lane&15 : K = [kk+8h .. +8)  and [kk+16+8h .. +8)
//   B frag, lane half h, col n=lane&15 : K = [kk+8h .. +8)  and [kk+16+8h .. +8)
//   C frag, VGPR rr : row = 8h + rr, col = lane&15
template<int ACT, int OUTBF>
__global__ __launch_bounds__(256)
void gemm_wmma_kernel(const __bf16* __restrict__ A, const __bf16* __restrict__ W,
                      const float* __restrict__ bias, void* __restrict__ C,
                      int M, int N, int K, int ldc)
{
    const int wid    = (int)((blockIdx.x * blockDim.x + threadIdx.x) >> 5);
    const int tilesN = N >> 6;
    const int total  = (M >> 5) * tilesN;
    if (wid >= total) return;
    const int tm   = wid / tilesN;
    const int tn   = wid - tm * tilesN;
    const int lane = threadIdx.x & 31;
    const int half = lane >> 4;
    const int r    = lane & 15;

    const __bf16* Arow0 = A + (size_t)(tm * 32 + r) * K;
    const __bf16* Arow1 = A + (size_t)(tm * 32 + 16 + r) * K;
    const __bf16* Wrow[4];
#pragma unroll
    for (int j = 0; j < 4; ++j)
        Wrow[j] = W + (size_t)(tn * 64 + j * 16 + r) * K;

    v8f acc[2][4];
#pragma unroll
    for (int i = 0; i < 2; ++i)
#pragma unroll
        for (int j = 0; j < 4; ++j)
#pragma unroll
            for (int e = 0; e < 8; ++e) acc[i][j][e] = 0.0f;

    const int a0 = half * 8;
    const int b0 = half * 8;

    // ping buffer: fragments for k = 0
    v16bf pa0 = load16(Arow0 + a0, Arow0 + 16 + a0);
    v16bf pa1 = load16(Arow1 + a0, Arow1 + 16 + a0);
    v16bf pb[4];
#pragma unroll
    for (int j = 0; j < 4; ++j)
        pb[j] = load16(Wrow[j] + b0, Wrow[j] + 16 + b0);

    v16bf qa0, qa1, qb[4];  // pong buffer

    for (int kk = 0; kk < K; kk += 64) {
        // ---- half 1: load pong @ kk+32 (always in-bounds), compute with ping ----
        const int k1 = kk + 32;
        if (k1 + 512 < K) __builtin_prefetch(Arow0 + k1 + 512, 0, 1);
        qa0 = load16(Arow0 + k1 + a0, Arow0 + k1 + 16 + a0);
        qa1 = load16(Arow1 + k1 + a0, Arow1 + k1 + 16 + a0);
#pragma unroll
        for (int j = 0; j < 4; ++j)
            qb[j] = load16(Wrow[j] + k1 + b0, Wrow[j] + k1 + 16 + b0);

#pragma unroll
        for (int j = 0; j < 4; ++j)
            acc[0][j] = __builtin_amdgcn_wmma_f32_16x16x32_bf16(
                            false, pa0, false, pb[j], (short)0, acc[0][j], false, false);
#pragma unroll
        for (int j = 0; j < 4; ++j)
            acc[1][j] = __builtin_amdgcn_wmma_f32_16x16x32_bf16(
                            false, pa1, false, pb[j], (short)0, acc[1][j], false, false);

        // ---- half 2: load ping @ kk+64 (clamped on last iter), compute with pong ----
        const int k2 = (kk + 64 < K) ? (kk + 64) : 0;
        pa0 = load16(Arow0 + k2 + a0, Arow0 + k2 + 16 + a0);
        pa1 = load16(Arow1 + k2 + a0, Arow1 + k2 + 16 + a0);
#pragma unroll
        for (int j = 0; j < 4; ++j)
            pb[j] = load16(Wrow[j] + k2 + b0, Wrow[j] + k2 + 16 + b0);

#pragma unroll
        for (int j = 0; j < 4; ++j)
            acc[0][j] = __builtin_amdgcn_wmma_f32_16x16x32_bf16(
                            false, qa0, false, qb[j], (short)0, acc[0][j], false, false);
#pragma unroll
        for (int j = 0; j < 4; ++j)
            acc[1][j] = __builtin_amdgcn_wmma_f32_16x16x32_bf16(
                            false, qa1, false, qb[j], (short)0, acc[1][j], false, false);
    }

#pragma unroll
    for (int i = 0; i < 2; ++i) {
#pragma unroll
        for (int j = 0; j < 4; ++j) {
            const int n  = tn * 64 + j * 16 + r;
            const float bn = bias ? bias[n] : 0.0f;
#pragma unroll
            for (int rr = 0; rr < 8; ++rr) {
                const int m = tm * 32 + i * 16 + half * 8 + rr;
                float v = acc[i][j][rr] + bn;
                if (ACT == 1) v = tanhf(v);
                const size_t idx = (size_t)m * (size_t)ldc + (size_t)n;
                if (OUTBF) ((__bf16*)C)[idx] = (__bf16)v;
                else       ((float*)C)[idx]  = v;
            }
        }
    }
}

// ---------------- elementwise kernels ----------------
__global__ void f32_to_bf16_kernel(const float* __restrict__ in, __bf16* __restrict__ out, int n) {
    int i = blockIdx.x * blockDim.x + threadIdx.x;
    if (i < n) out[i] = (__bf16)in[i];
}

__global__ void copy_f32_kernel(const float* __restrict__ in, float* __restrict__ out, int n) {
    int i = blockIdx.x * blockDim.x + threadIdx.x;
    if (i < n) out[i] = in[i];
}

// scores[b][s] = mask ? sum_h v[h]*tanh(pk[b][s][h] + qp[b][h]) : -1e9   (one wave per (b,s))
__global__ __launch_bounds__(256)
void attn_scores_kernel(const __bf16* __restrict__ pk, const float* __restrict__ qp,
                        const float* __restrict__ v, const unsigned char* __restrict__ mask,
                        float* __restrict__ scores)
{
    const int wid = (int)((blockIdx.x * blockDim.x + threadIdx.x) >> 5);
    if (wid >= BB * SS) return;
    const int b = wid / SS, s = wid - b * SS;
    const int lane = threadIdx.x & 31;
    const __bf16* row = pk + ((size_t)b * SS + s) * HH;
    const float*  q   = qp + (size_t)b * HH;
    float sum = 0.0f;
#pragma unroll 4
    for (int h = lane; h < HH; h += 32)
        sum += v[h] * tanhf((float)row[h] + q[h]);
#pragma unroll
    for (int off = 16; off > 0; off >>= 1) sum += __shfl_xor(sum, off, 32);
    if (lane == 0) scores[wid] = mask[b * SS + s] ? sum : -1.0e9f;
}

// row softmax over S=256: one 256-thread block per batch row
__global__ __launch_bounds__(256)
void softmax_kernel(const float* __restrict__ scores, float* __restrict__ alphas)
{
    __shared__ float red[256];
    const int b = blockIdx.x, s = threadIdx.x;
    const float x = scores[b * SS + s];
    red[s] = x; __syncthreads();
#pragma unroll
    for (int off = 128; off > 0; off >>= 1) {
        if (s < off) red[s] = fmaxf(red[s], red[s + off]);
        __syncthreads();
    }
    const float mx = red[0]; __syncthreads();
    const float e = __expf(x - mx);
    red[s] = e; __syncthreads();
#pragma unroll
    for (int off = 128; off > 0; off >>= 1) {
        if (s < off) red[s] += red[s + off];
        __syncthreads();
    }
    alphas[b * SS + s] = e / red[0];
}

// context[b][d] = sum_s alphas[b][s] * enc_bf16[b][s][d]   grid (H2/256, B)
__global__ __launch_bounds__(256)
void context_kernel(const float* __restrict__ alphas, const __bf16* __restrict__ enc,
                    float* __restrict__ ctx)
{
    __shared__ float al[SS];
    const int b = blockIdx.y;
    const int d = blockIdx.x * 256 + threadIdx.x;
    al[threadIdx.x] = alphas[b * SS + threadIdx.x];
    __syncthreads();
    const __bf16* base = enc + (size_t)b * SS * H2 + d;
    float sum = 0.0f;
#pragma unroll 4
    for (int s = 0; s < SS; ++s) sum = fmaf(al[s], (float)base[(size_t)s * H2], sum);
    ctx[b * H2 + d] = sum;
}

// x_bf16(B,1280) = [embed_t | context];  also fills embed/context slots of cat_po(B,1792)
__global__ void build_x_kernel(const float* __restrict__ trg_embed, const float* __restrict__ ctx,
                               __bf16* __restrict__ x, __bf16* __restrict__ cat, int t)
{
    const int id = blockIdx.x * blockDim.x + threadIdx.x;
    if (id >= BB * KIH) return;
    const int b = id / KIH, i = id - b * KIH;
    float v;
    if (i < EE) {
        v = trg_embed[(size_t)b * TT * EE + (size_t)t * EE + i];
        cat[(size_t)b * PP + i] = (__bf16)v;                 // [0,256): embed
    } else {
        v = ctx[b * H2 + (i - EE)];
        cat[(size_t)b * PP + 512 + i] = (__bf16)v;           // [768,1792): context
    }
    x[(size_t)b * KIH + i] = (__bf16)v;
}

// GRU pointwise; writes new hidden (f32+bf16), output slot of cat_po, decoder_states
__global__ void gru_kernel(const float* __restrict__ gi, const float* __restrict__ gh,
                           float* __restrict__ hidden, __bf16* __restrict__ hidden_bf,
                           __bf16* __restrict__ cat, float* __restrict__ out_states, int t)
{
    const int id = blockIdx.x * blockDim.x + threadIdx.x;
    if (id >= BB * HH) return;
    const int b = id / HH, j = id - b * HH;
    const float* gib = gi + (size_t)b * G3;
    const float* ghb = gh + (size_t)b * G3;
    const float r = sigmoidf_(gib[j]        + ghb[j]);
    const float z = sigmoidf_(gib[HH + j]   + ghb[HH + j]);
    const float n = tanhf(gib[2 * HH + j] + r * ghb[2 * HH + j]);
    const float h  = hidden[id];
    const float hn = (1.0f - z) * n + z * h;
    hidden[id]    = hn;
    hidden_bf[id] = (__bf16)hn;
    cat[(size_t)b * PP + EE + j] = (__bf16)hn;               // [256,768): output
    out_states[(size_t)b * TT * HH + (size_t)t * HH + j] = hn;
}

// ---------------- host side ----------------
static inline void gemm(hipStream_t st, int act, int outbf,
                        const __bf16* A, const __bf16* W, const float* bias,
                        void* C, int M, int N, int K, int ldc)
{
    const int waves  = (M / 32) * (N / 64);
    const int blocks = (waves + 7) / 8;
    if (act == 1)      gemm_wmma_kernel<1, 0><<<blocks, 256, 0, st>>>(A, W, bias, C, M, N, K, ldc);
    else if (outbf)    gemm_wmma_kernel<0, 1><<<blocks, 256, 0, st>>>(A, W, bias, C, M, N, K, ldc);
    else               gemm_wmma_kernel<0, 0><<<blocks, 256, 0, st>>>(A, W, bias, C, M, N, K, ldc);
}

static inline void conv_bf16(hipStream_t st, const float* in, __bf16* out, int n) {
    f32_to_bf16_kernel<<<(n + 255) / 256, 256, 0, st>>>(in, out, n);
}

extern "C" void kernel_launch(void* const* d_in, const int* in_sizes, int n_in,
                              void* d_out, int out_size, void* d_ws, size_t ws_size,
                              hipStream_t stream)
{
    (void)in_sizes; (void)n_in; (void)out_size; (void)ws_size;
    // inputs (setup_inputs order)
    const float* trg_embed      = (const float*)d_in[0];   // (B,T,E)
    const float* encoder_hidden = (const float*)d_in[1];   // (B,S,2H)
    const float* encoder_final  = (const float*)d_in[2];   // (1,B,2H)
    const unsigned char* src_mask = (const unsigned char*)d_in[3]; // (B,1,S) bool
    const float* W_key    = (const float*)d_in[5];   // (H,2H)
    const float* W_query  = (const float*)d_in[6];   // (H,H)
    const float* v_energy = (const float*)d_in[7];   // (1,H)
    const float* W_ih     = (const float*)d_in[8];   // (3H, E+2H)
    const float* W_hh     = (const float*)d_in[9];   // (3H, H)
    const float* b_ih     = (const float*)d_in[10];
    const float* b_hh     = (const float*)d_in[11];
    const float* W_bridge = (const float*)d_in[12];  // (H,2H)
    const float* b_bridge = (const float*)d_in[13];
    const float* W_po     = (const float*)d_in[14];  // (H, H+2H+E)

    float* out = (float*)d_out;
    float* out_states = out;                               // (B,T,H)
    float* out_hfinal = out + (size_t)BB * TT * HH;        // (1,B,H)
    float* out_pre    = out_hfinal + (size_t)BB * HH;      // (B,T,H)

    // workspace bump allocator (256B aligned)
    char* ws = (char*)d_ws; size_t cur = 0;
    auto alloc = [&](size_t bytes) -> char* {
        char* p = ws + cur;
        cur = (cur + bytes + 255) & ~(size_t)255;
        return p;
    };
    __bf16* pk_bf    = (__bf16*)alloc((size_t)BB * SS * HH * 2);   // proj_key bf16
    __bf16* enc_bf   = (__bf16*)alloc((size_t)BB * SS * H2 * 2);   // encoder_hidden bf16
    __bf16* wkey_bf  = (__bf16*)alloc((size_t)HH * H2 * 2);
    __bf16* wq_bf    = (__bf16*)alloc((size_t)HH * HH * 2);
    __bf16* wih_bf   = (__bf16*)alloc((size_t)G3 * KIH * 2);
    __bf16* whh_bf   = (__bf16*)alloc((size_t)G3 * HH * 2);
    __bf16* wpo_bf   = (__bf16*)alloc((size_t)HH * PP * 2);
    __bf16* wbr_bf   = (__bf16*)alloc((size_t)HH * H2 * 2);
    __bf16* ef_bf    = (__bf16*)alloc((size_t)BB * H2 * 2);
    float*  hidden   = (float*)alloc((size_t)BB * HH * 4);
    __bf16* hid_bf   = (__bf16*)alloc((size_t)BB * HH * 2);
    float*  qproj    = (float*)alloc((size_t)BB * HH * 4);
    float*  scores   = (float*)alloc((size_t)BB * SS * 4);
    float*  alphas   = (float*)alloc((size_t)BB * SS * 4);
    float*  ctx      = (float*)alloc((size_t)BB * H2 * 4);
    __bf16* x_bf     = (__bf16*)alloc((size_t)BB * KIH * 2);
    float*  gi       = (float*)alloc((size_t)BB * G3 * 4);
    float*  gh       = (float*)alloc((size_t)BB * G3 * 4);
    __bf16* cat_bf   = (__bf16*)alloc((size_t)BB * PP * 2);

    // ---- setup: bf16 conversions ----
    conv_bf16(stream, encoder_hidden, enc_bf, BB * SS * H2);
    conv_bf16(stream, W_key,    wkey_bf, HH * H2);
    conv_bf16(stream, W_query,  wq_bf,   HH * HH);
    conv_bf16(stream, W_ih,     wih_bf,  G3 * KIH);
    conv_bf16(stream, W_hh,     whh_bf,  G3 * HH);
    conv_bf16(stream, W_po,     wpo_bf,  HH * PP);
    conv_bf16(stream, W_bridge, wbr_bf,  HH * H2);
    conv_bf16(stream, encoder_final, ef_bf, BB * H2);

    // ---- bridge: hidden0 = tanh(encoder_final @ W_bridge^T + b_bridge) ----
    gemm(stream, /*act=tanh*/1, 0, ef_bf, wbr_bf, b_bridge, hidden, BB, HH, H2, HH);
    conv_bf16(stream, hidden, hid_bf, BB * HH);

    // ---- proj_key = encoder_hidden @ W_key^T  (bf16 output, stays L2-resident) ----
    gemm(stream, 0, /*outbf=*/1, enc_bf, wkey_bf, nullptr, pk_bf, BB * SS, HH, H2, HH);

    // ---- 64 sequential decoder steps ----
    for (int t = 0; t < TT; ++t) {
        // query projection: qproj = hidden @ W_query^T
        gemm(stream, 0, 0, hid_bf, wq_bf, nullptr, qproj, BB, HH, HH, HH);
        // attention scores (one wave per (b,s))
        attn_scores_kernel<<<(BB * SS * 32 + 255) / 256, 256, 0, stream>>>(
            pk_bf, qproj, v_energy, src_mask, scores);
        // softmax over S
        softmax_kernel<<<BB, 256, 0, stream>>>(scores, alphas);
        // context = alphas @ encoder_hidden
        context_kernel<<<dim3(H2 / 256, BB), 256, 0, stream>>>(alphas, enc_bf, ctx);
        // x = [embed_t | context] (bf16) + embed/context slots of cat_po
        build_x_kernel<<<(BB * KIH + 255) / 256, 256, 0, stream>>>(trg_embed, ctx, x_bf, cat_bf, t);
        // GRU gates
        gemm(stream, 0, 0, x_bf,   wih_bf, b_ih, gi, BB, G3, KIH, G3);
        gemm(stream, 0, 0, hid_bf, whh_bf, b_hh, gh, BB, G3, HH,  G3);
        // GRU pointwise: updates hidden, writes decoder_states[:,t,:], fills cat_po output slot
        gru_kernel<<<(BB * HH + 255) / 256, 256, 0, stream>>>(
            gi, gh, hidden, hid_bf, cat_bf, out_states, t);
        // pre_output[:,t,:] = cat_po @ W_po^T   (strided C: ldc = T*H, base offset t*H)
        gemm(stream, 0, 0, cat_bf, wpo_bf, nullptr, out_pre + (size_t)t * HH,
             BB, HH, PP, TT * HH);
    }

    // hidden_final = final hidden state
    copy_f32_kernel<<<(BB * HH + 255) / 256, 256, 0, stream>>>(hidden, out_hfinal, BB * HH);
}
